// MoMEGate_58583353917581
// MI455X (gfx1250) — matrix operand order repaired and checked by reading, not compiled, not run on hardware
//
#include <hip/hip_runtime.h>
#include <hip/hip_bf16.h>

#define HIDDEN  4096
#define NSHARD  64
#define BROWS   16384
#define KSTEP   32

typedef __attribute__((ext_vector_type(16))) __bf16 v16bf;
typedef __attribute__((ext_vector_type(8)))  float  v8f;

// ---- pre-pass: weight fp32 (64 x 4096) -> bf16 row-major in workspace ----
__global__ __launch_bounds__(256)
void cvt_weight_bf16(const float* __restrict__ w, __bf16* __restrict__ o, int n) {
    int i = blockIdx.x * blockDim.x + threadIdx.x;
    if (i < n) o[i] = (__bf16)w[i];
}

__device__ __forceinline__ v16bf cvt_a(float4 a0, float4 a1, float4 a2, float4 a3) {
    v16bf a;
    a[0]  = (__bf16)a0.x; a[1]  = (__bf16)a0.y; a[2]  = (__bf16)a0.z; a[3]  = (__bf16)a0.w;
    a[4]  = (__bf16)a1.x; a[5]  = (__bf16)a1.y; a[6]  = (__bf16)a1.z; a[7]  = (__bf16)a1.w;
    a[8]  = (__bf16)a2.x; a[9]  = (__bf16)a2.y; a[10] = (__bf16)a2.z; a[11] = (__bf16)a2.w;
    a[12] = (__bf16)a3.x; a[13] = (__bf16)a3.y; a[14] = (__bf16)a3.z; a[15] = (__bf16)a3.w;
    return a;
}

// ---- fused gate: bf16 WMMA GEMM (rows x 64 experts) + top2 + 2-way softmax ----
// kdim passed at runtime so the k-loop trip count is NOT a compile-time
// constant: keeps the steady-state loop compact and I$-resident instead of
// a 40+KB fully-unrolled instruction stream.
__global__ __launch_bounds__(128, 1)
void moe_gate_kernel(const float* __restrict__ h,
                     const __bf16* __restrict__ wb,
                     const float* __restrict__ bias,
                     float* __restrict__ out_w,
                     int*   __restrict__ out_i,
                     int kdim) {
    __shared__ float lds_logits[64][65];   // 64 rows per block, stride 65 -> conflict-free

    const int tid   = threadIdx.x;
    const int wave  = tid >> 5;            // 4 waves per block
    const int lane  = tid & 31;
    const int l16   = lane & 15;
    const int lhalf = lane >> 4;

    const int rowBase = blockIdx.x * 64 + wave * 16;   // 16 rows per wave

    v8f acc0 = {}, acc1 = {}, acc2 = {}, acc3 = {};

    const float*  arow = h  + (size_t)(rowBase + l16) * HIDDEN;
    const __bf16* bcol = wb + (size_t)l16 * HIDDEN;

    // ---- software-pipelined main loop: loads for step k+1 issued while
    //      WMMAs consume step k ----
    float4 a0, a1, a2, a3;        // raw fp32 A chunk (next)
    v16bf  nb0, nb1, nb2, nb3;    // bf16 B tiles (next)

#define LOAD_A(K0)                                                       \
    {   const float* ap_ = arow + (K0) + lhalf * 16;                     \
        a0 = *(const float4*)(ap_ + 0);                                  \
        a1 = *(const float4*)(ap_ + 4);                                  \
        a2 = *(const float4*)(ap_ + 8);                                  \
        a3 = *(const float4*)(ap_ + 12);                                 \
        __builtin_prefetch(ap_ + 96, 0, 1); }
#define LOAD_B(K0)                                                       \
    {   const __bf16* bp_ = bcol + (K0) + lhalf * 16;                    \
        nb0 = *(const v16bf*)(bp_);                                      \
        nb1 = *(const v16bf*)(bp_ + 16 * HIDDEN);                        \
        nb2 = *(const v16bf*)(bp_ + 32 * HIDDEN);                        \
        nb3 = *(const v16bf*)(bp_ + 48 * HIDDEN); }

    LOAD_A(0);
    LOAD_B(0);

    #pragma unroll 2
    for (int k0 = 0; k0 < kdim - KSTEP; k0 += KSTEP) {
        // consume current values into private regs
        v16bf a  = cvt_a(a0, a1, a2, a3);
        v16bf b0 = nb0, b1 = nb1, b2 = nb2, b3 = nb3;
        // issue next iteration's loads (waited on at top of next iteration)
        LOAD_A(k0 + KSTEP);
        LOAD_B(k0 + KSTEP);
        // 4 WMMAs cover the load latency
        acc0 = __builtin_amdgcn_wmma_f32_16x16x32_bf16(false, a, false, b0, (short)0, acc0, false, false);
        acc1 = __builtin_amdgcn_wmma_f32_16x16x32_bf16(false, a, false, b1, (short)0, acc1, false, false);
        acc2 = __builtin_amdgcn_wmma_f32_16x16x32_bf16(false, a, false, b2, (short)0, acc2, false, false);
        acc3 = __builtin_amdgcn_wmma_f32_16x16x32_bf16(false, a, false, b3, (short)0, acc3, false, false);
    }
    {   // epilogue: last k-step
        v16bf a = cvt_a(a0, a1, a2, a3);
        acc0 = __builtin_amdgcn_wmma_f32_16x16x32_bf16(false, a, false, nb0, (short)0, acc0, false, false);
        acc1 = __builtin_amdgcn_wmma_f32_16x16x32_bf16(false, a, false, nb1, (short)0, acc1, false, false);
        acc2 = __builtin_amdgcn_wmma_f32_16x16x32_bf16(false, a, false, nb2, (short)0, acc2, false, false);
        acc3 = __builtin_amdgcn_wmma_f32_16x16x32_bf16(false, a, false, nb3, (short)0, acc3, false, false);
    }
#undef LOAD_A
#undef LOAD_B

    // ---- spill logits to LDS: C layout VGPR r -> M = r + 8*lhalf, N = l16 ----
    #pragma unroll
    for (int r = 0; r < 8; ++r) {
        const int m = wave * 16 + r + lhalf * 8;
        lds_logits[m][ 0 + l16] = acc0[r];
        lds_logits[m][16 + l16] = acc1[r];
        lds_logits[m][32 + l16] = acc2[r];
        lds_logits[m][48 + l16] = acc3[r];
    }
    __syncthreads();

    // ---- per-row top-2 + renormalize (full softmax cancels analytically) ----
    if (tid < 64) {
        const int grow = blockIdx.x * 64 + tid;
        float v1 = -INFINITY, v2 = -INFINITY;
        int   i1 = 0, i2 = 0;
        #pragma unroll 4
        for (int j = 0; j < NSHARD; ++j) {
            float v = lds_logits[tid][j] + bias[j];
            if (v > v1)      { v2 = v1; i2 = i1; v1 = v; i1 = j; }
            else if (v > v2) { v2 = v;  i2 = j; }
        }
        float e   = __expf(v2 - v1);        // <= 1
        float inv = __frcp_rn(1.0f + e);
        out_w[grow * 2 + 0] = inv;
        out_w[grow * 2 + 1] = e * inv;
        out_i[grow * 2 + 0] = i1;
        out_i[grow * 2 + 1] = i2;
    }
}

extern "C" void kernel_launch(void* const* d_in, const int* in_sizes, int n_in,
                              void* d_out, int out_size, void* d_ws, size_t ws_size,
                              hipStream_t stream) {
    const float* h    = (const float*)d_in[0];
    const float* w    = (const float*)d_in[1];
    const float* bias = (const float*)d_in[2];

    __bf16* wb    = (__bf16*)d_ws;                 // 64*4096*2 = 512 KB scratch
    float*  out_w = (float*)d_out;                 // (B,2) fp32
    int*    out_i = (int*)(out_w + 2 * BROWS);     // (B,2) int32, flat-concatenated

    const int nw = NSHARD * HIDDEN;
    cvt_weight_bf16<<<(nw + 255) / 256, 256, 0, stream>>>(w, wb, nw);
    moe_gate_kernel<<<BROWS / 64, 128, 0, stream>>>(h, wb, bias, out_w, out_i, HIDDEN);
}
